// Method_GCN_25907242729542
// MI455X (gfx1250) — matrix operand order repaired and checked by reading, not compiled, not run on hardware
//
#include <hip/hip_runtime.h>
#include <hip/hip_bf16.h>

typedef _Float16 v16h __attribute__((ext_vector_type(16)));
typedef float    v8f  __attribute__((ext_vector_type(8)));

#define C1 16
#define C2 3
#define KPAD 512                 // padded K (feat=500 -> 16 k-blocks of 32)
#define NKB  (KPAD / 32)

// ---------------- degree / norm ----------------

__global__ void k_init_deg(float* __restrict__ deg, int N) {
    int i = blockIdx.x * blockDim.x + threadIdx.x;
    if (i < N) deg[i] = 1.0f;  // self-loop contribution
}

__global__ void k_deg_edges(const long long* __restrict__ dst,
                            float* __restrict__ deg, long long E) {
    long long e = (long long)blockIdx.x * blockDim.x + threadIdx.x;
    if (e < E) atomicAdd(&deg[(int)dst[e]], 1.0f);
}

__global__ void k_dinv(const float* __restrict__ deg,
                       float* __restrict__ dinv, int N) {
    int i = blockIdx.x * blockDim.x + threadIdx.x;
    if (i < N) {
        float d = deg[i];
        dinv[i] = (d > 0.0f) ? rsqrtf(d) : 0.0f;
    }
}

// ---------------- GEMM1: h1 = x @ W1 via WMMA f16->f32 ----------------
// One wave computes a 16(M) x 16(N) f32 tile; K tiled by 32 in f16.
// W1 is pre-swizzled into LDS in per-(kblock,lane) order so each lane's
// 16 B-halves are one contiguous 32-byte chunk.

__global__ __launch_bounds__(256)
void k_gemm1_wmma(const float* __restrict__ x, const float* __restrict__ W1,
                  float* __restrict__ h1, int N, int feat) {
    __shared__ alignas(32) _Float16 w1s[NKB * 32 * 16];   // 16 KB

    int tid = threadIdx.x;
    // swizzled preload: idx = ((kb*32 + lane)*16 + v)
    // lane -> hi = lane>>4, n = lane&15 ; element = W1[(kb*32 + (v&8)*2 + hi*8 + (v&7))*16 + n]
    for (int idx = tid; idx < NKB * 32 * 16; idx += 256) {
        int v    = idx & 15;
        int lane = (idx >> 4) & 31;
        int kb   = idx >> 9;
        int k    = kb * 32 + (v & 8) * 2 + (lane >> 4) * 8 + (v & 7);
        int n    = lane & 15;
        w1s[idx] = (k < feat) ? (_Float16)W1[k * C1 + n] : (_Float16)0.0f;
    }
    __syncthreads();

    int wave = tid >> 5;
    int lane = tid & 31;
    int hi   = lane >> 4;        // which 16-lane half
    int n    = lane & 15;        // output column
    int mbase = blockIdx.x * 128 + wave * 16;

    int arow = mbase + (lane & 15);
    size_t rbase = (size_t)((arow < N) ? arow : (N - 1)) * (size_t)feat;
    const float* xr = x + rbase;

    v8f c = {};
    int fullkb = feat >> 5;                      // 15 full blocks
    for (int kb = 0; kb < fullkb; ++kb) {
        int k0 = kb << 5;
        // A: lane holds row (lane&15); halves 0..7  -> K = k0+hi*8 .. +7
        //                              halves 8..15 -> K = k0+16+hi*8 .. +7
        const float4* p0 = (const float4*)(xr + k0 + hi * 8);
        const float4* p1 = (const float4*)(xr + k0 + 16 + hi * 8);
        float4 q0 = p0[0], q1 = p0[1];
        float4 r0 = p1[0], r1 = p1[1];

        v16h a;
        a[0]  = (_Float16)q0.x; a[1]  = (_Float16)q0.y;
        a[2]  = (_Float16)q0.z; a[3]  = (_Float16)q0.w;
        a[4]  = (_Float16)q1.x; a[5]  = (_Float16)q1.y;
        a[6]  = (_Float16)q1.z; a[7]  = (_Float16)q1.w;
        a[8]  = (_Float16)r0.x; a[9]  = (_Float16)r0.y;
        a[10] = (_Float16)r0.z; a[11] = (_Float16)r0.w;
        a[12] = (_Float16)r1.x; a[13] = (_Float16)r1.y;
        a[14] = (_Float16)r1.z; a[15] = (_Float16)r1.w;

        v16h b = *(const v16h*)&w1s[(kb * 32 + lane) * 16];

        c = __builtin_amdgcn_wmma_f32_16x16x32_f16(
                false, a, false, b, (short)0, c, false, false);
    }

    // tail block (feat % 32 != 0): clamped unconditional loads + selects
    if (feat & 31) {
        int k0 = fullkb << 5;
        v16h a;
#pragma unroll
        for (int j = 0; j < 8; ++j) {
            int ka = k0 + hi * 8 + j;
            int kc = k0 + 16 + hi * 8 + j;
            float va = xr[(ka < feat) ? ka : 0];
            float vc = xr[(kc < feat) ? kc : 0];
            a[j]     = (ka < feat) ? (_Float16)va : (_Float16)0.0f;
            a[8 + j] = (kc < feat) ? (_Float16)vc : (_Float16)0.0f;
        }
        v16h b = *(const v16h*)&w1s[(fullkb * 32 + lane) * 16];
        c = __builtin_amdgcn_wmma_f32_16x16x32_f16(
                false, a, false, b, (short)0, c, false, false);
    }

    // D layout: VGPR r -> M = r + 8*hi, column = n
#pragma unroll
    for (int r = 0; r < 8; ++r) {
        int m = mbase + r + 8 * hi;
        if (m < N) h1[(size_t)m * C1 + n] = c[r];
    }
}

// ---------------- layer-1 aggregation ----------------

__global__ void k_init_agg1(const float* __restrict__ h1,
                            const float* __restrict__ dinv,
                            float* __restrict__ agg1, int N) {
    int t = blockIdx.x * blockDim.x + threadIdx.x;
    if (t < N * C1) {
        int i = t >> 4;
        float w = dinv[i];
        agg1[t] = h1[t] * w * w;   // self-loop term
    }
}

__global__ void k_scatter16(const long long* __restrict__ src,
                            const long long* __restrict__ dst,
                            const float* __restrict__ dinv,
                            const float* __restrict__ h,
                            float* __restrict__ agg, long long E) {
    long long t = (long long)blockIdx.x * blockDim.x + threadIdx.x;
    if (t >= E * C1) return;
    long long e = t >> 4;
    int c = (int)(t & 15);
    int s = (int)src[e];
    int d = (int)dst[e];
    float w = dinv[s] * dinv[d];
    atomicAdd(&agg[(size_t)d * C1 + c], h[(size_t)s * C1 + c] * w);
}

// ---------------- relu + GEMM2 (16->3) + self-loop init of agg2 ----------------

__global__ void k_relu_gemm2(const float* __restrict__ agg1,
                             const float* __restrict__ b1,
                             const float* __restrict__ W2,
                             const float* __restrict__ dinv,
                             float* __restrict__ h2,
                             float* __restrict__ agg2, int N) {
    int i = blockIdx.x * blockDim.x + threadIdx.x;
    if (i >= N) return;
    float t[C1];
#pragma unroll
    for (int c = 0; c < C1; ++c) {
        float v = agg1[(size_t)i * C1 + c] + b1[c];
        t[c] = v > 0.0f ? v : 0.0f;
    }
    float w = dinv[i];
    float w2 = w * w;
#pragma unroll
    for (int j = 0; j < C2; ++j) {
        float acc = 0.0f;
#pragma unroll
        for (int c = 0; c < C1; ++c) acc += t[c] * W2[c * C2 + j];
        h2[(size_t)i * C2 + j]   = acc;
        agg2[(size_t)i * C2 + j] = acc * w2;   // self-loop term
    }
}

__global__ void k_scatter3(const long long* __restrict__ src,
                           const long long* __restrict__ dst,
                           const float* __restrict__ dinv,
                           const float* __restrict__ h2,
                           float* __restrict__ agg2, long long E) {
    long long e = (long long)blockIdx.x * blockDim.x + threadIdx.x;
    if (e >= E) return;
    int s = (int)src[e];
    int d = (int)dst[e];
    float w = dinv[s] * dinv[d];
#pragma unroll
    for (int j = 0; j < C2; ++j)
        atomicAdd(&agg2[(size_t)d * C2 + j], h2[(size_t)s * C2 + j] * w);
}

// ---------------- bias + log_softmax ----------------

__global__ void k_logsoftmax(const float* __restrict__ agg2,
                             const float* __restrict__ b2,
                             float* __restrict__ out, int N) {
    int i = blockIdx.x * blockDim.x + threadIdx.x;
    if (i >= N) return;
    float z[C2];
    float m = -3.4e38f;
#pragma unroll
    for (int j = 0; j < C2; ++j) {
        z[j] = agg2[(size_t)i * C2 + j] + b2[j];
        m = fmaxf(m, z[j]);
    }
    float s = 0.0f;
#pragma unroll
    for (int j = 0; j < C2; ++j) s += __expf(z[j] - m);
    float lse = m + __logf(s);
#pragma unroll
    for (int j = 0; j < C2; ++j) out[(size_t)i * C2 + j] = z[j] - lse;
}

// ---------------- host launcher ----------------

extern "C" void kernel_launch(void* const* d_in, const int* in_sizes, int n_in,
                              void* d_out, int out_size, void* d_ws, size_t ws_size,
                              hipStream_t stream) {
    const float*     x   = (const float*)d_in[0];
    const long long* ei  = (const long long*)d_in[1];
    const float*     W1  = (const float*)d_in[2];
    const float*     b1  = (const float*)d_in[3];
    const float*     W2  = (const float*)d_in[4];
    const float*     b2  = (const float*)d_in[5];
    float*           out = (float*)d_out;

    const int  c1   = in_sizes[3];            // 16
    const int  feat = in_sizes[2] / c1;       // 500
    const int  N    = in_sizes[0] / feat;     // 100000
    const long long E = (long long)in_sizes[1] / 2;  // 3200000
    (void)c1;

    const long long* src = ei;
    const long long* dst = ei + E;

    // workspace layout (floats): deg[N] dinv[N] h1[16N] agg1[16N] h2[3N] agg2[3N]
    float* ws   = (float*)d_ws;
    float* deg  = ws;
    float* dinv = deg  + N;
    float* h1   = dinv + N;
    float* agg1 = h1   + (size_t)N * C1;
    float* h2   = agg1 + (size_t)N * C1;
    float* agg2 = h2   + (size_t)N * C2;

    const int T = 256;

    k_init_deg<<<(N + T - 1) / T, T, 0, stream>>>(deg, N);
    k_deg_edges<<<(unsigned)((E + T - 1) / T), T, 0, stream>>>(dst, deg, E);
    k_dinv<<<(N + T - 1) / T, T, 0, stream>>>(deg, dinv, N);

    k_gemm1_wmma<<<(N + 127) / 128, T, 0, stream>>>(x, W1, h1, N, feat);

    k_init_agg1<<<((size_t)N * C1 + T - 1) / T, T, 0, stream>>>(h1, dinv, agg1, N);
    k_scatter16<<<(unsigned)((E * C1 + T - 1) / T), T, 0, stream>>>(src, dst, dinv, h1, agg1, E);

    k_relu_gemm2<<<(N + T - 1) / T, T, 0, stream>>>(agg1, b1, W2, dinv, h2, agg2, N);
    k_scatter3<<<(unsigned)((E + T - 1) / T), T, 0, stream>>>(src, dst, dinv, h2, agg2, E);

    k_logsoftmax<<<(N + T - 1) / T, T, 0, stream>>>(agg2, b2, out, N);
}
